// AttentionWeights_32298154066180
// MI455X (gfx1250) — matrix operand order repaired and checked by reading, not compiled, not run on hardware
//
#include <hip/hip_runtime.h>

// ---------------------------------------------------------------------------
// Types for CDNA5 WMMA (wave32): bf16 A/B (16 x bf16 per lane), f32 C/D (8 x f32)
// ---------------------------------------------------------------------------
typedef __attribute__((ext_vector_type(16))) __bf16 v16bf;
typedef __attribute__((ext_vector_type(8)))  __bf16 v8bf;
typedef __attribute__((ext_vector_type(8)))  float  v8f;

#define DIM_B 8
#define DIM_N 2048
#define DIM_D 256
#define BN    (DIM_B * DIM_N)   // 16384 rows total

// ---------------------------------------------------------------------------
// Kernel 0: transpose + convert Wq/Wk (f32, DxD row-major) into bf16 W^T.
// W^T row c = column c of W, so the WMMA B-operand load becomes a contiguous
// 16-element run per lane.
// ---------------------------------------------------------------------------
__global__ void wt_convert_k(const float* __restrict__ Wq,
                             const float* __restrict__ Wk,
                             __bf16* __restrict__ WqT,
                             __bf16* __restrict__ WkT) {
    int idx = blockIdx.x * blockDim.x + threadIdx.x;   // 0 .. D*D-1
    if (idx >= DIM_D * DIM_D) return;
    int d = idx >> 8;          // row of W
    int e = idx & (DIM_D - 1); // col of W
    WqT[e * DIM_D + d] = (__bf16)Wq[idx];
    WkT[e * DIM_D + d] = (__bf16)Wk[idx];
}

// ---------------------------------------------------------------------------
// Kernel 1: Y = relu(X @ W + bias) in bf16.  One wave per 16-row x 64-col strip.
// A (16x32 bf16) per-lane layout used here (contiguous K, same order as B):
//   lane = hf*16 + lr : row lr, elements [0..15] = K kb+hf*16 .. +15 (contiguous)
// B (32x16 bf16): lane holds column lr, contiguous K run kb+hf*16 .. +15  -> one
//   32B vector load from W^T row (col).
// ---------------------------------------------------------------------------
__global__ __launch_bounds__(32) void proj_relu_bf16_k(
    const float*  __restrict__ X,     // BN x D, f32
    const __bf16* __restrict__ WT,    // D x D, bf16 (transposed)
    const float*  __restrict__ bias,  // D
    __bf16*       __restrict__ Y)     // BN x D, bf16
{
    const int rt   = blockIdx.x;      // 0..BN/16-1 : 16-row tile
    const int cg   = blockIdx.y;      // 0..3       : 64-col group (4 tiles)
    const int lane = threadIdx.x;
    const int lr   = lane & 15;
    const int hf   = lane >> 4;
    const int row  = rt * 16 + lr;

    v8f acc[4] = {};
    for (int kk = 0; kk < 8; ++kk) {          // K = 256 in steps of 32
        const int kb = kk * 32;
        const float* xp = X + (size_t)row * DIM_D + kb + hf * 16;
        v16bf a;
        #pragma unroll
        for (int i = 0; i < 8; ++i) {
            a[i]     = (__bf16)xp[i];
            a[8 + i] = (__bf16)xp[8 + i];
        }
        #pragma unroll
        for (int j = 0; j < 4; ++j) {
            const int col = cg * 64 + j * 16 + lr;
            v16bf b = *(const v16bf*)(WT + (size_t)col * DIM_D + kb + hf * 16);
            acc[j] = __builtin_amdgcn_wmma_f32_16x16x32_bf16(
                false, a, false, b, (short)0, acc[j], false, false);
        }
    }
    // C/D layout: VGPR r holds (M = r + 8*hf, N = lr)
    #pragma unroll
    for (int j = 0; j < 4; ++j) {
        const int gcol = cg * 64 + j * 16 + lr;
        const float bv = bias[gcol];
        #pragma unroll
        for (int r = 0; r < 8; ++r) {
            const int grow = rt * 16 + r + 8 * hf;
            float v = acc[j][r] + bv;
            Y[(size_t)grow * DIM_D + gcol] = (__bf16)(v > 0.f ? v : 0.f);
        }
    }
}

// ---------------------------------------------------------------------------
// Kernel 2: scores + sigmoid + symmetrize + zero-diagonal, fused.
// One wave per upper-triangular pair of 32x32 super-tiles (NT <= MT):
//   acc1[i][j] = Q[NT,i-th 16] . K[MT,j-th 16]^T   (8 x wmma_bf16 over K=256)
//   acc2[j][i] = Q[MT,j-th 16] . K[NT,i-th 16]^T
// Epilogue: sig = sigmoid(acc/16); out(n,m) = 0.5*(S1 + S2^T), out(m,n) mirror,
// diagonal zeroed.  Each 134MB output element is written exactly once
// (diagonal super-tiles write idempotently from the same wave).
// ---------------------------------------------------------------------------
__global__ __launch_bounds__(32) void scores_sym_k(
    const __bf16* __restrict__ Q,    // B x N x D bf16
    const __bf16* __restrict__ K,    // B x N x D bf16
    float*        __restrict__ out)  // B x N x N f32
{
    const int NT = blockIdx.x;               // 32-row super-tile
    const int MT = blockIdx.y;               // 32-col super-tile
    if (NT > MT) return;                     // upper triangle only
    const int bb   = blockIdx.z;
    const int lane = threadIdx.x;
    const int lr   = lane & 15;
    const int hf   = lane >> 4;

    const __bf16* Qb = Q + (size_t)bb * DIM_N * DIM_D;
    const __bf16* Kb = K + (size_t)bb * DIM_N * DIM_D;
    float*        ob = out + (size_t)bb * DIM_N * DIM_N;

    v8f acc1[2][2] = {};   // [i][j] : rows NT*32+i*16, cols MT*32+j*16
    v8f acc2[2][2] = {};   // [j][i] : rows MT*32+j*16, cols NT*32+i*16

    for (int kk = 0; kk < 8; ++kk) {
        const int kb = kk * 32;
        v16bf aN[2], aM[2], bM[2], bN[2];
        #pragma unroll
        for (int t = 0; t < 2; ++t) {
            // A operands (row tiles of Q): contiguous 16-element K run kb+hf*16.. (matches B)
            const __bf16* pn = Qb + (size_t)(NT * 32 + t * 16 + lr) * DIM_D + kb + hf * 16;
            v8bf nlo = *(const v8bf*)pn;
            v8bf nhi = *(const v8bf*)(pn + 8);
            aN[t] = __builtin_shufflevector(nlo, nhi, 0,1,2,3,4,5,6,7,8,9,10,11,12,13,14,15);

            const __bf16* pm = Qb + (size_t)(MT * 32 + t * 16 + lr) * DIM_D + kb + hf * 16;
            v8bf mlo = *(const v8bf*)pm;
            v8bf mhi = *(const v8bf*)(pm + 8);
            aM[t] = __builtin_shufflevector(mlo, mhi, 0,1,2,3,4,5,6,7,8,9,10,11,12,13,14,15);

            // B operands: B[k][c] = K[row c][k] -> lane holds row of K, contiguous 16
            bM[t] = *(const v16bf*)(Kb + (size_t)(MT * 32 + t * 16 + lr) * DIM_D + kb + hf * 16);
            bN[t] = *(const v16bf*)(Kb + (size_t)(NT * 32 + t * 16 + lr) * DIM_D + kb + hf * 16);
        }
        #pragma unroll
        for (int i = 0; i < 2; ++i)
            #pragma unroll
            for (int j = 0; j < 2; ++j) {
                acc1[i][j] = __builtin_amdgcn_wmma_f32_16x16x32_bf16(
                    false, aN[i], false, bM[j], (short)0, acc1[i][j], false, false);
                acc2[j][i] = __builtin_amdgcn_wmma_f32_16x16x32_bf16(
                    false, aM[j], false, bN[i], (short)0, acc2[j][i], false, false);
            }
    }

    // scale by 1/sqrt(D)=1/16 and sigmoid, in place
    #pragma unroll
    for (int i = 0; i < 2; ++i)
        #pragma unroll
        for (int j = 0; j < 2; ++j)
            #pragma unroll
            for (int r = 0; r < 8; ++r) {
                acc1[i][j][r] = 1.f / (1.f + __expf(acc1[i][j][r] * -0.0625f));
                acc2[i][j][r] = 1.f / (1.f + __expf(acc2[i][j][r] * -0.0625f));
            }

    // symmetrize each 16x16 sub-tile pair through LDS transpose
    __shared__ float t1[256];
    __shared__ float t2[256];
    #pragma unroll 1
    for (int i = 0; i < 2; ++i) {
        #pragma unroll 1
        for (int j = 0; j < 2; ++j) {
            #pragma unroll
            for (int r = 0; r < 8; ++r) {
                t1[(r + 8 * hf) * 16 + lr] = acc1[i][j][r];   // S1[i][j] tile
                t2[(r + 8 * hf) * 16 + lr] = acc2[j][i][r];   // S2[j][i] tile
            }
            __syncthreads();
            #pragma unroll
            for (int r = 0; r < 8; ++r) {
                const int tr   = r + 8 * hf;                 // row within tile
                const int grow = NT * 32 + i * 16 + tr;      // n
                const int gcol = MT * 32 + j * 16 + lr;      // m
                float wp = 0.5f * (acc1[i][j][r] + t2[lr * 16 + tr]);
                if (grow == gcol) wp = 0.f;
                ob[(size_t)grow * DIM_N + gcol] = wp;

                const int mrow = MT * 32 + j * 16 + tr;
                const int mcol = NT * 32 + i * 16 + lr;
                float wm = 0.5f * (acc2[j][i][r] + t1[lr * 16 + tr]);
                if (mrow == mcol) wm = 0.f;
                ob[(size_t)mrow * DIM_N + mcol] = wm;
            }
            __syncthreads();
        }
    }
}

// ---------------------------------------------------------------------------
// Host-side launcher.  Inputs: X, Wq, bq, Wk, bk (all f32).
// Workspace layout (bytes):
//   [0,        8 MiB)  q  bf16 (BN x D)
//   [8 MiB,   16 MiB)  k  bf16 (BN x D)
//   [16 MiB, +128KiB)  Wq^T bf16
//   [+256KiB, ...)     Wk^T bf16           total ~16.5 MiB
// ---------------------------------------------------------------------------
extern "C" void kernel_launch(void* const* d_in, const int* in_sizes, int n_in,
                              void* d_out, int out_size, void* d_ws, size_t ws_size,
                              hipStream_t stream) {
    const float* X  = (const float*)d_in[0];
    const float* Wq = (const float*)d_in[1];
    const float* bq = (const float*)d_in[2];
    const float* Wk = (const float*)d_in[3];
    const float* bk = (const float*)d_in[4];
    float* out = (float*)d_out;

    char* ws = (char*)d_ws;
    __bf16* q   = (__bf16*)(ws);
    __bf16* k   = (__bf16*)(ws + (size_t)8 * 1024 * 1024);
    __bf16* WqT = (__bf16*)(ws + (size_t)16 * 1024 * 1024);
    __bf16* WkT = (__bf16*)(ws + (size_t)16 * 1024 * 1024 + 256 * 1024);

    // K0: W -> W^T bf16
    wt_convert_k<<<(DIM_D * DIM_D) / 256, 256, 0, stream>>>(Wq, Wk, WqT, WkT);

    // K1: q = relu(X Wq + bq), k = relu(X Wk + bk)   (bf16 out)
    proj_relu_bf16_k<<<dim3(BN / 16, 4), 32, 0, stream>>>(X, WqT, bq, q);
    proj_relu_bf16_k<<<dim3(BN / 16, 4), 32, 0, stream>>>(X, WkT, bk, k);

    // K2: fused scores + sigmoid + symmetrize + zero-diag
    scores_sym_k<<<dim3(DIM_N / 32, DIM_N / 32, DIM_B), 32, 0, stream>>>(q, k, out);
}
